// SigmoidAttention_14267881357727
// MI455X (gfx1250) — compile-verified, matrix operand order checked by reading, not run on hardware
//
#include <hip/hip_runtime.h>
#include <hip/hip_bf16.h>

// ---------------------------------------------------------------------------
// Sigmoid GQA attention for MI455X (gfx1250), bf16 WMMA pipeline + async TDM.
//   B=1, S=4096, FEATURE=2048, H=16, KVH=4, D=128
// Pipeline:
//   1) fp32 -> bf16 conversion of activations + weights
//   2) WMMA NT-GEMM (32Mx64N wave tiles): Q/K/V projections (bf16 out)
//   3) WMMA attention: K/V tiles double-buffered in LDS, prefetched by the
//      Tensor Data Mover one block ahead (TENSORcnt-pipelined),
//      S = Q K^T, sigmoid(scale*S - log S + causal), O = P V
//   4) WMMA NT-GEMM: output projection (f32 out)
// ---------------------------------------------------------------------------

#define SEQ  4096
#define FD   2048
#define QD   2048
#define KVD  512
#define HD   128

typedef __attribute__((ext_vector_type(16))) __bf16       v16bf;
typedef __attribute__((ext_vector_type(8)))  float        v8f;
typedef __attribute__((ext_vector_type(4)))  unsigned int u32x4;
typedef __attribute__((ext_vector_type(4)))  int          i32x4;
typedef __attribute__((ext_vector_type(8)))  int          i32x8;

union Frag16 {
    v16bf bf;
    u32x4 q[2];
};

#if __has_builtin(__builtin_amdgcn_tensor_load_to_lds) && \
    __has_builtin(__builtin_amdgcn_s_wait_tensorcnt)
#define USE_TDM 1
#else
#define USE_TDM 0
#endif

__device__ __forceinline__ unsigned short f32_to_bf16(float f) {
    unsigned int u = __float_as_uint(f);
    unsigned int r = u + 0x7FFFu + ((u >> 16) & 1u);   // round to nearest even
    return (unsigned short)(r >> 16);
}

// fp32 -> bf16 elementwise
__global__ void to_bf16_kernel(const float* __restrict__ src,
                               unsigned short* __restrict__ dst, int n) {
    int i = blockIdx.x * blockDim.x + threadIdx.x;
    if (i < n) dst[i] = f32_to_bf16(src[i]);
}

#if USE_TDM
// ---------------------------------------------------------------------------
// 2D Tensor Data Mover load: tile [tile_d1 rows x tile_d0 elems] of 2-byte
// data from a row-major tensor (row stride = stride0 elems) into LDS.
// D# layout per CDNA5 ISA ch.8 (group0: count/lds/global/type, group1: dims).
// ---------------------------------------------------------------------------
__device__ __forceinline__ void tdm_load_2d_b16(
    unsigned lds_off, const void* gaddr,
    unsigned tensor_d0, unsigned tensor_d1,
    unsigned tile_d0, unsigned tile_d1, unsigned stride0)
{
    unsigned long long ga = (unsigned long long)(size_t)gaddr;
    u32x4 g0;
    g0.x = 1u;                                     // count=1 (valid), user mode
    g0.y = lds_off;                                // lds_addr (bytes)
    g0.z = (unsigned)(ga & 0xFFFFFFFFu);           // global_addr[31:0]
    g0.w = (unsigned)((ga >> 32) & 0x01FFFFFFu)    // global_addr[56:32]
         | 0x80000000u;                            // type=2 ("image"), bit127
    i32x8 g1;
    g1[0] = 0x00010000;                            // wg_mask=0, data_size=1(2B)
    g1[1] = (int)(tensor_d0 << 16);                // tensor_dim0[15:0] @bits63:48
    g1[2] = (int)((tensor_d0 >> 16) | (tensor_d1 << 16));
    g1[3] = (int)((tensor_d1 >> 16) | (tile_d0 << 16));   // tile_dim0 @127:112
    g1[4] = (int)(tile_d1 & 0xFFFFu);              // tile_dim1; tile_dim2=0 (2D)
    g1[5] = (int)stride0;                          // tensor_dim0_stride[31:0]
    g1[6] = 0;                                     // stride0[47:32], stride1 lo
    g1[7] = 0;                                     // stride1 hi
    i32x4 z4 = {0, 0, 0, 0};
#if defined(__clang_major__) && (__clang_major__ >= 23)
    i32x8 z8 = {0, 0, 0, 0, 0, 0, 0, 0};
    __builtin_amdgcn_tensor_load_to_lds(g0, g1, z4, z4, z8, 0);
#else
    __builtin_amdgcn_tensor_load_to_lds(g0, g1, z4, z4, 0);
#endif
}
#endif

// ---------------------------------------------------------------------------
// NT-GEMM: C[M,N] = A[M,K] * W[N,K]^T + bias[N]
// Block: 256 threads = 8 waves as 2(M) x 4(N); wave tile 32M x 64N.
// Block tile: 64M x 256N. Both operands K-contiguous -> 16B fragment loads.
// ---------------------------------------------------------------------------
template <bool BF16OUT>
__global__ __launch_bounds__(256) void gemm_nt_kernel(
    const unsigned short* __restrict__ A,
    const unsigned short* __restrict__ W,
    const float* __restrict__ bias,
    void* __restrict__ outp,
    int M, int N, int Kd)
{
    const int tid  = threadIdx.x;
    const int w    = tid >> 5;
    const int lane = tid & 31;
    const int lrow = lane & 15;
    const int kh   = lane >> 4;

    const int m0 = blockIdx.x * 64  + (w & 1) * 32;
    const int n0 = blockIdx.y * 256 + (w >> 1) * 64;

    const v8f zero = {0.f, 0.f, 0.f, 0.f, 0.f, 0.f, 0.f, 0.f};
    v8f acc[2][4];
#pragma unroll
    for (int mi = 0; mi < 2; ++mi)
#pragma unroll
        for (int t = 0; t < 4; ++t) acc[mi][t] = zero;

    const unsigned short* arow0 = A + (size_t)(m0 + lrow) * Kd;
    const unsigned short* arow1 = arow0 + (size_t)16 * Kd;

    for (int k0 = 0; k0 < Kd; k0 += 32) {
        Frag16 a0, a1;
        {
            const unsigned short* p0 = arow0 + k0 + kh * 8;
            const unsigned short* p1 = arow1 + k0 + kh * 8;
            a0.q[0] = *(const u32x4*)(p0);
            a0.q[1] = *(const u32x4*)(p0 + 16);
            a1.q[0] = *(const u32x4*)(p1);
            a1.q[1] = *(const u32x4*)(p1 + 16);
            __builtin_prefetch(p0 + 64, 0, 1);
        }
#pragma unroll
        for (int t = 0; t < 4; ++t) {
            const unsigned short* wp =
                W + (size_t)(n0 + t * 16 + lrow) * Kd + k0 + kh * 8;
            Frag16 bf_;
            bf_.q[0] = *(const u32x4*)(wp);
            bf_.q[1] = *(const u32x4*)(wp + 16);
            acc[0][t] = __builtin_amdgcn_wmma_f32_16x16x32_bf16(
                false, a0.bf, false, bf_.bf, (short)0, acc[0][t], false, false);
            acc[1][t] = __builtin_amdgcn_wmma_f32_16x16x32_bf16(
                false, a1.bf, false, bf_.bf, (short)0, acc[1][t], false, false);
        }
    }

#pragma unroll
    for (int mi = 0; mi < 2; ++mi) {
#pragma unroll
        for (int t = 0; t < 4; ++t) {
#pragma unroll
            for (int j = 0; j < 8; ++j) {
                const int m = m0 + mi * 16 + j + 8 * kh;
                const int n = n0 + t * 16 + lrow;
                const float val = acc[mi][t][j] + bias[n];
                if (BF16OUT)
                    ((unsigned short*)outp)[(size_t)m * N + n] = f32_to_bf16(val);
                else
                    ((float*)outp)[(size_t)m * N + n] = val;
            }
        }
    }
}

// ---------------------------------------------------------------------------
// Attention: sigmoid(scale*QK^T - log S + causal) @ V, GQA 16q/4kv heads.
// Grid: (32 q-blocks of 128, 16 heads). Block: 256 threads = 8 waves;
// wave owns 16 query rows. Key blocks of 64. K/V tiles double-buffered in
// LDS; the Tensor Data Mover prefetches block kb+1 while block kb computes
// (fallback: cooperative vector copies). V transposed LDS->LDS for B-frags.
// ---------------------------------------------------------------------------
__global__ __launch_bounds__(256) void attn_kernel(
    const unsigned short* __restrict__ Q,   // [SEQ][QD] bf16
    const unsigned short* __restrict__ K,   // [SEQ][KVD] bf16
    const unsigned short* __restrict__ V,   // [SEQ][KVD] bf16
    unsigned short* __restrict__ O)         // [SEQ][QD] bf16
{
    __shared__ __attribute__((aligned(16))) unsigned short Krow[2][64][128]; // 32K
    __shared__ __attribute__((aligned(16))) unsigned short Vrow[2][64][128]; // 32K
    __shared__ __attribute__((aligned(16))) unsigned short Vt[128][72];      // 18K
    __shared__ __attribute__((aligned(16))) unsigned short Ps[8][16][40];    // 10K

    const int qb   = blockIdx.x;       // query block, 0..31
    const int h    = blockIdx.y;       // head, 0..15
    const int g    = h >> 2;           // kv head (groups = 4)
    const int tid  = threadIdx.x;
    const int w    = tid >> 5;
    const int lane = tid & 31;
    const int lrow = lane & 15;
    const int kh   = lane >> 4;

    const float scale = 0.08838834764831845f;   // 1/sqrt(128)
    const float sbias = -8.317766166719343f;    // -log(4096)

    const int qrow = qb * 128 + w * 16;

    // ---- cache Q fragments (16 rows x 128 head-dim -> 4 chunks of 32) ----
    v16bf qf[4];
    {
        const unsigned short* qbase = Q + (size_t)(qrow + lrow) * QD + h * HD;
#pragma unroll
        for (int kk = 0; kk < 4; ++kk) {
            Frag16 f;
            const unsigned short* p = qbase + kk * 32 + kh * 8;
            f.q[0] = *(const u32x4*)(p);
            f.q[1] = *(const u32x4*)(p + 16);
            qf[kk] = f.bf;
        }
    }

    const v8f zero = {0.f, 0.f, 0.f, 0.f, 0.f, 0.f, 0.f, 0.f};
    v8f oacc[8];
#pragma unroll
    for (int t = 0; t < 8; ++t) oacc[t] = zero;

    const int kb_end = 2 * qb + 1;     // 64-wide key blocks, causal bound

#if USE_TDM
    // prologue: TDM-prefetch the first K/V tile pair into buffer 0
    if (w == 0) {
        tdm_load_2d_b16((unsigned)(size_t)&Krow[0][0][0],
                        K + (size_t)0 * KVD + g * HD, KVD, SEQ, 128, 64, KVD);
        tdm_load_2d_b16((unsigned)(size_t)&Vrow[0][0][0],
                        V + (size_t)0 * KVD + g * HD, KVD, SEQ, 128, 64, KVD);
    }
#endif

    for (int kb = 0; kb <= kb_end; ++kb) {
        const int cur = kb & 1;
#if USE_TDM
        if (w == 0)
            __builtin_amdgcn_s_wait_tensorcnt(0);   // buffer[cur] landed
        __syncthreads();                            // visible to all waves;
                                                    // prior block's reads done
        if (w == 0 && kb < kb_end) {                // async prefetch kb+1
            const int nb = cur ^ 1;
            tdm_load_2d_b16((unsigned)(size_t)&Krow[nb][0][0],
                            K + (size_t)((kb + 1) * 64) * KVD + g * HD,
                            KVD, SEQ, 128, 64, KVD);
            tdm_load_2d_b16((unsigned)(size_t)&Vrow[nb][0][0],
                            V + (size_t)((kb + 1) * 64) * KVD + g * HD,
                            KVD, SEQ, 128, 64, KVD);
        }
#else
        __syncthreads();
        {
            const int r  = tid >> 2;   // 0..63
            const int qd = tid & 3;    // 32-wide column quarter
            const unsigned short* kp =
                K + (size_t)(kb * 64 + r) * KVD + g * HD + qd * 32;
            const unsigned short* vp =
                V + (size_t)(kb * 64 + r) * KVD + g * HD + qd * 32;
#pragma unroll
            for (int c = 0; c < 4; ++c) {
                *(u32x4*)&Krow[cur][r][qd * 32 + c * 8] =
                    *(const u32x4*)(kp + c * 8);
                *(u32x4*)&Vrow[cur][r][qd * 32 + c * 8] =
                    *(const u32x4*)(vp + c * 8);
            }
        }
        __syncthreads();
#endif

        // ---- transpose Vrow[cur] -> Vt[d][k_local] (LDS -> LDS) ----
        {
            const int r  = tid >> 2;   // key row 0..63
            const int qd = tid & 3;    // 32-wide chunk of head dim
#pragma unroll
            for (int c8 = 0; c8 < 4; ++c8) {
                u32x4 d = *(const u32x4*)&Vrow[cur][r][qd * 32 + c8 * 8];
                const unsigned short* ds = (const unsigned short*)&d;
                const int cbase = qd * 32 + c8 * 8;
#pragma unroll
                for (int e = 0; e < 8; ++e)
                    Vt[cbase + e][r] = ds[e];
            }
        }
        __syncthreads();

        for (int chunk = 0; chunk < 2; ++chunk) {
            // ---- two 16x16 score tiles: keys [kb*64+chunk*32, +32) ----
#pragma unroll
            for (int half = 0; half < 2; ++half) {
                const int nt   = chunk * 2 + half;
                const int kcol = kb * 64 + nt * 16;
                // preload the 4 K fragments, then run the WMMA chain
                Frag16 kf[4];
#pragma unroll
                for (int kk = 0; kk < 4; ++kk) {
                    const unsigned short* p =
                        &Krow[cur][nt * 16 + lrow][kk * 32 + kh * 8];
                    kf[kk].q[0] = *(const u32x4*)(p);
                    kf[kk].q[1] = *(const u32x4*)(p + 16);
                }
                v8f s = zero;
#pragma unroll
                for (int kk = 0; kk < 4; ++kk)
                    s = __builtin_amdgcn_wmma_f32_16x16x32_bf16(
                        false, qf[kk], false, kf[kk].bf, (short)0, s,
                        false, false);
                // sigmoid + causal mask, bf16 into wave-local P scratch
                const int ki = kcol + lrow;
#pragma unroll
                for (int j = 0; j < 8; ++j) {
                    const int m  = j + 8 * kh;
                    const int qi = qrow + m;
                    const float x  = s[j] * scale + sbias;
                    const float pv =
                        (ki <= qi) ? (1.0f / (1.0f + __expf(-x))) : 0.0f;
                    Ps[w][m][half * 16 + lrow] = f32_to_bf16(pv);
                }
            }
            __builtin_amdgcn_wave_barrier();   // LDS in-order within wave

            // ---- P A-fragment + preloaded V B-fragments, 8 WMMA chain ----
            Frag16 pf;
            {
                const unsigned short* p = &Ps[w][lrow][kh * 8];
                pf.q[0] = *(const u32x4*)(p);
                pf.q[1] = *(const u32x4*)(p + 16);
            }
            Frag16 vf[8];
#pragma unroll
            for (int t = 0; t < 8; ++t) {
                const unsigned short* p =
                    &Vt[t * 16 + lrow][chunk * 32 + kh * 8];
                vf[t].q[0] = *(const u32x4*)(p);
                vf[t].q[1] = *(const u32x4*)(p + 16);
            }
#pragma unroll
            for (int t = 0; t < 8; ++t)
                oacc[t] = __builtin_amdgcn_wmma_f32_16x16x32_bf16(
                    false, pf.bf, false, vf[t].bf, (short)0, oacc[t],
                    false, false);
            __builtin_amdgcn_wave_barrier();
        }
    }

    // ---- store O tile (bf16, heads concatenated) ----
#pragma unroll
    for (int t = 0; t < 8; ++t) {
#pragma unroll
        for (int j = 0; j < 8; ++j) {
            const int m = j + 8 * kh;
            O[(size_t)(qrow + m) * QD + h * HD + t * 16 + lrow] =
                f32_to_bf16(oacc[t][j]);
        }
    }
}

// ---------------------------------------------------------------------------
extern "C" void kernel_launch(void* const* d_in, const int* in_sizes, int n_in,
                              void* d_out, int out_size, void* d_ws, size_t ws_size,
                              hipStream_t stream) {
    const float* hs = (const float*)d_in[0];
    // d_in[1] = attention_mask (causal; implemented analytically)
    const float* Wq = (const float*)d_in[2];
    const float* bq = (const float*)d_in[3];
    const float* Wk = (const float*)d_in[4];
    const float* bk = (const float*)d_in[5];
    const float* Wv = (const float*)d_in[6];
    const float* bv = (const float*)d_in[7];
    const float* Wo = (const float*)d_in[8];
    const float* bo = (const float*)d_in[9];

    char*  ws  = (char*)d_ws;
    size_t off = 0;
    auto alloc = [&](size_t bytes) -> unsigned short* {
        unsigned short* p = (unsigned short*)(ws + off);
        off += (bytes + 255) & ~(size_t)255;
        return p;
    };

    unsigned short* Xb  = alloc((size_t)SEQ * FD  * 2);   // hidden bf16
    unsigned short* Wqb = alloc((size_t)QD  * FD  * 2);
    unsigned short* Wkb = alloc((size_t)KVD * FD  * 2);
    unsigned short* Wvb = alloc((size_t)KVD * FD  * 2);
    unsigned short* Wob = alloc((size_t)FD  * FD  * 2);
    unsigned short* Qb  = alloc((size_t)SEQ * QD  * 2);
    unsigned short* Kb  = alloc((size_t)SEQ * KVD * 2);
    unsigned short* Vb  = alloc((size_t)SEQ * KVD * 2);
    unsigned short* Ab  = alloc((size_t)SEQ * QD  * 2);   // attention out bf16

    auto cvt = [&](const float* src, unsigned short* dst, int n) {
        to_bf16_kernel<<<(n + 255) / 256, 256, 0, stream>>>(src, dst, n);
    };
    cvt(hs, Xb,  SEQ * FD);
    cvt(Wq, Wqb, QD  * FD);
    cvt(Wk, Wkb, KVD * FD);
    cvt(Wv, Wvb, KVD * FD);
    cvt(Wo, Wob, FD  * FD);

    // projections (bf16 out)
    gemm_nt_kernel<true><<<dim3(SEQ / 64, QD  / 256), 256, 0, stream>>>(
        Xb, Wqb, bq, Qb, SEQ, QD, FD);
    gemm_nt_kernel<true><<<dim3(SEQ / 64, KVD / 256), 256, 0, stream>>>(
        Xb, Wkb, bk, Kb, SEQ, KVD, FD);
    gemm_nt_kernel<true><<<dim3(SEQ / 64, KVD / 256), 256, 0, stream>>>(
        Xb, Wvb, bv, Vb, SEQ, KVD, FD);

    // sigmoid attention
    attn_kernel<<<dim3(SEQ / 128, 16), 256, 0, stream>>>(Qb, Kb, Vb, Ab);

    // output projection (f32 out)
    gemm_nt_kernel<false><<<dim3(SEQ / 64, FD / 256), 256, 0, stream>>>(
        Ab, Wob, bo, d_out, SEQ, FD, FD);
}
